// ModulatedDeformConv2d_8684423872827
// MI455X (gfx1250) — matrix operand order repaired and checked by reading, not compiled, not run on hardware
//
#include <hip/hip_runtime.h>

// ---------------------------------------------------------------------------
// Modulated deformable conv2d (DCNv2), fused deform-im2col + bf16 WMMA GEMM.
// B=8, C=Cout=64, H=W=128, 3x3, stride 1, pad 1, dil 1.
// ---------------------------------------------------------------------------

#define BATCH 8
#define CIN   64
#define COUT  64
#define H     128
#define W     128
#define HW    (H * W)
#define K2    9
#define Q     (CIN * K2)      // 576 = GEMM K
#define QPAD  584             // LDS stride in halves (16B aligned, conflict pad)
#define TILE_PIX 32           // pixels per workgroup (one row segment)

typedef __attribute__((ext_vector_type(16))) __bf16 v16bf;
typedef __attribute__((ext_vector_type(8)))  float  v8f;

union Frag {
    v16bf v;
    uint4 q[2];
};

__device__ __forceinline__ unsigned short f2bf(float f) {
    unsigned int u = __float_as_uint(f);
    // round-to-nearest-even fp32 -> bf16
    unsigned int r = (u + 0x7FFFu + ((u >> 16) & 1u)) >> 16;
    return (unsigned short)r;
}

// ---- Stage 0: weight fp32 -> bf16, flat [Cout][Cin*9] order -----------------
__global__ void wconv_kernel(const float* __restrict__ w,
                             unsigned short* __restrict__ o, int n) {
    int i = blockIdx.x * 256 + threadIdx.x;
    if (i < n) o[i] = f2bf(w[i]);
}

// ---- Main fused kernel ------------------------------------------------------
__global__ void __launch_bounds__(256)
dcn_fused_kernel(const float* __restrict__ x,
                 const float* __restrict__ offset,
                 const float* __restrict__ mask,
                 const unsigned short* __restrict__ wbf,
                 const float* __restrict__ bias,
                 float* __restrict__ out) {
    __shared__ unsigned short cols[TILE_PIX * QPAD];   // bf16 col tile  (36.5 KB)
    __shared__ int4   eoff[TILE_PIX * K2];             // 4 corner pixel indices
    __shared__ float4 ewt [TILE_PIX * K2];             // 4 corner weights (mask folded)

    const int tid  = threadIdx.x;
    const int wv   = tid >> 5;          // wave id 0..7
    const int lane = tid & 31;

    const int blk     = blockIdx.x;     // 0..4095
    const int b       = blk >> 9;       // 512 tiles per image
    const int pixbase = (blk & 511) * TILE_PIX;
    const int h       = pixbase >> 7;   // row (tiles never wrap: 128/32=4)
    const int wc      = pixbase & 127;  // col base

    // ---------- Stage 1: per (pixel, tap) bilinear setup ----------
    for (int e = tid; e < TILE_PIX * K2; e += 256) {
        const int p  = e & 31;
        const int k  = e >> 5;
        const int ww = wc + p;
        const int hw = h * W + ww;

        const float dx = offset[(b * 2 * K2 + 2 * k    ) * HW + hw];
        const float dy = offset[(b * 2 * K2 + 2 * k + 1) * HW + hw];
        const float m  = mask  [(b * K2 + k) * HW + hw];

        const float py = (float)(h - 1 + (k / 3)) + dy;
        const float px = (float)(ww - 1 + (k % 3)) + dx;

        const float y0f = floorf(py), x0f = floorf(px);
        const int   y0 = (int)y0f,    x0 = (int)x0f;
        const float ly = py - y0f,    lx = px - x0f;
        const float hy = 1.f - ly,    hx = 1.f - lx;
        const int   y1 = y0 + 1,      x1 = x0 + 1;

        const float vy0 = (y0 >= 0 && y0 < H) ? 1.f : 0.f;
        const float vy1 = (y1 >= 0 && y1 < H) ? 1.f : 0.f;
        const float vx0 = (x0 >= 0 && x0 < W) ? 1.f : 0.f;
        const float vx1 = (x1 >= 0 && x1 < W) ? 1.f : 0.f;

        const int y0c = min(max(y0, 0), H - 1), y1c = min(max(y1, 0), H - 1);
        const int x0c = min(max(x0, 0), W - 1), x1c = min(max(x1, 0), W - 1);

        eoff[e] = make_int4(y0c * W + x0c, y0c * W + x1c,
                            y1c * W + x0c, y1c * W + x1c);
        ewt[e]  = make_float4(hy * hx * m * vy0 * vx0,
                              hy * lx * m * vy0 * vx1,
                              ly * hx * m * vy1 * vx0,
                              ly * lx * m * vy1 * vx1);
    }
    __syncthreads();

    // ---------- Stage 2: gather channels -> bf16 col tile in LDS ----------
    // lane = pixel (coalesced gathers along W); wave strides over (c,k) pairs.
    {
        const float* xb = x + (size_t)b * CIN * HW;
        const int p = lane;
        for (int j = wv; j < CIN * K2; j += 8) {       // 72 iters per wave
            const int c = j / 9;
            const int k = j - c * 9;
            const int e = (k << 5) + p;
            const int4   o4 = eoff[e];
            const float4 w4 = ewt[e];
            const float* xc = xb + c * HW;
            const float  v  = w4.x * xc[o4.x] + w4.y * xc[o4.y] +
                              w4.z * xc[o4.z] + w4.w * xc[o4.w];
            cols[p * QPAD + c * 9 + k] = f2bf(v);
        }
    }
    __syncthreads();

    // ---------- Stage 3: WMMA GEMM (16o x 16n per wave, K=576) ----------
    const int obase = (wv & 3) * 16;    // 4 o-tiles
    const int nb    = (wv >> 2) * 16;   // 2 n-tiles
    const int n16   = lane & 15;
    const int half  = lane >> 4;

    v8f acc = {};
    for (int ks = 0; ks < Q; ks += 32) {
        Frag A, Bf;
        // A: row M = n16; halves hold K = ks+8*half..+7 and ks+16+8*half..+7
        const unsigned short* ap = wbf + (obase + n16) * Q + ks + 8 * half;
        A.q[0] = *(const uint4*)(ap);
        A.q[1] = *(const uint4*)(ap + 16);
        // B: col N = n16; halves hold K = ks+8*half..+7 and ks+16+8*half..+7
        const unsigned short* bp = &cols[(nb + n16) * QPAD + ks + 8 * half];
        Bf.q[0] = *(const uint4*)(bp);
        Bf.q[1] = *(const uint4*)(bp + 16);

        acc = __builtin_amdgcn_wmma_f32_16x16x32_bf16(
            false, A.v, false, Bf.v, (short)0, acc, false, false);
    }

    // ---------- Store: D vgpr v <-> M = v + 8*half, N = lane&15 ----------
    const int pix = pixbase + nb + n16;
    #pragma unroll
    for (int v = 0; v < 8; ++v) {
        const int o = obase + v + 8 * half;
        out[((size_t)(b * COUT + o)) * HW + pix] = acc[v] + bias[o];
    }
}

// ---------------------------------------------------------------------------
extern "C" void kernel_launch(void* const* d_in, const int* in_sizes, int n_in,
                              void* d_out, int out_size, void* d_ws, size_t ws_size,
                              hipStream_t stream) {
    (void)in_sizes; (void)n_in; (void)out_size; (void)ws_size;
    const float* x      = (const float*)d_in[0];
    const float* offset = (const float*)d_in[1];
    const float* mask   = (const float*)d_in[2];
    const float* weight = (const float*)d_in[3];
    const float* bias   = (const float*)d_in[4];
    float* out = (float*)d_out;
    unsigned short* wbf = (unsigned short*)d_ws;   // 64*576*2 = 73728 B

    const int nw = COUT * Q;
    wconv_kernel<<<(nw + 255) / 256, 256, 0, stream>>>(weight, wbf, nw);

    dim3 grid(BATCH * HW / TILE_PIX);   // 4096 workgroups
    dcn_fused_kernel<<<grid, 256, 0, stream>>>(x, offset, mask, wbf, bias, out);
}
